// EquiMLP_6708738916905
// MI455X (gfx1250) — compile-verified, matrix operand order checked by reading, not run on hardware
//
#include <hip/hip_runtime.h>
#include <stdint.h>

// ---------------------------------------------------------------------------
// Problem constants (from reference): B=16, N_CGS=128, N_ATOMS=1024, KNN=8,
// W=1024, E=N_CGS*KNN=1024 edges/batch, BE=B*E=16384.
// ---------------------------------------------------------------------------
#define BATCH     16
#define NCGS      128
#define NATOMS    1024
#define KNN       8
#define WID       1024
#define EDGES     (NCGS * KNN)        // 1024
#define BE        (BATCH * EDGES)     // 16384

typedef __attribute__((ext_vector_type(16))) __bf16 v16bf;
typedef __attribute__((ext_vector_type(8)))  float  v8f;

union FragBF {
    uint4 q[2];   // two 16-byte LDS reads
    v16bf v;      // WMMA operand
};

__device__ __forceinline__ unsigned short f2bf(float f) {
    unsigned int u = __float_as_uint(f);
    unsigned int r = u + 0x7FFFu + ((u >> 16) & 1u);   // round-to-nearest-even
    return (unsigned short)(r >> 16);
}
__device__ __forceinline__ float bf2f(unsigned short h) {
    return __uint_as_float(((unsigned int)h) << 16);
}

// CDNA5 async data mover: 16B global -> LDS per lane, tracked by ASYNCcnt.
__device__ __forceinline__ void async_g2l_b128(unsigned int lds_off,
                                               const void* gptr) {
    asm volatile("global_load_async_to_lds_b128 %0, %1, off"
                 :: "v"(lds_off), "v"(gptr)
                 : "memory");
}
__device__ __forceinline__ void wait_async0() {
    asm volatile("s_wait_asynccnt 0x0" ::: "memory");
}

// ---------------------------------------------------------------------------
// 0) Transpose + convert weights to bf16 (N-major so B fragments are
//    contiguous 16B runs in LDS).  i covers 1024*1024 elements.
// ---------------------------------------------------------------------------
__global__ __launch_bounds__(256) void wt_kernel(const float* __restrict__ w1,
                                                 const float* __restrict__ w2,
                                                 unsigned short* __restrict__ w1T,
                                                 unsigned short* __restrict__ w2T) {
    int i = blockIdx.x * 256 + threadIdx.x;      // 0 .. 1048575
    int k = i >> 10;
    int n = i & 1023;
    w1T[n * WID + k] = f2bf(w1[i]);
    w2T[n * WID + k] = f2bf(w2[i]);
}

// ---------------------------------------------------------------------------
// 1) kNN (8 nearest excluding self) + edge vectors + edge lengths.
//    One block per batch, one thread per CG bead.
// ---------------------------------------------------------------------------
__global__ __launch_bounds__(128) void knn_kernel(const float* __restrict__ cg_xyz,
                                                  float* __restrict__ dist_vec,
                                                  float* __restrict__ dist) {
    __shared__ float pts[NCGS * 3];
    const int b = blockIdx.x;
    const int c = threadIdx.x;
    for (int i = c; i < NCGS * 3; i += 128) pts[i] = cg_xyz[b * NCGS * 3 + i];
    __syncthreads();

    const float px = pts[c * 3 + 0], py = pts[c * 3 + 1], pz = pts[c * 3 + 2];
    float bestd[KNN];
    int   besti[KNN];
#pragma unroll
    for (int s = 0; s < KNN; ++s) { bestd[s] = 3.4e38f; besti[s] = 0; }

    for (int j = 0; j < NCGS; ++j) {
        if (j == c) continue;
        float ddx = pts[j * 3 + 0] - px;
        float ddy = pts[j * 3 + 1] - py;
        float ddz = pts[j * 3 + 2] - pz;
        float d2 = ddx * ddx + ddy * ddy + ddz * ddz;
        if (d2 < bestd[KNN - 1]) {
            int s = KNN - 1;
            while (s > 0 && bestd[s - 1] > d2) {
                bestd[s] = bestd[s - 1]; besti[s] = besti[s - 1]; --s;
            }
            bestd[s] = d2; besti[s] = j;
        }
    }
#pragma unroll
    for (int s = 0; s < KNN; ++s) {
        int ge = (b * NCGS + c) * KNN + s;       // global edge id
        int j  = besti[s];
        dist_vec[ge * 3 + 0] = pts[j * 3 + 0] - px;
        dist_vec[ge * 3 + 1] = pts[j * 3 + 1] - py;
        dist_vec[ge * 3 + 2] = pts[j * 3 + 2] - pz;
        dist[ge] = sqrtf(bestd[s]);
    }
}

// ---------------------------------------------------------------------------
// 2) h0[e,j] = relu(dist[e]*w0[j] + b0[j])  ->  bf16
// ---------------------------------------------------------------------------
__global__ __launch_bounds__(256) void h0_kernel(const float* __restrict__ dist,
                                                 const float* __restrict__ w0,
                                                 const float* __restrict__ b0,
                                                 unsigned short* __restrict__ h0) {
    const int e = blockIdx.x;                     // 0..BE-1
    const float d = dist[e];
    unsigned short* row = h0 + (size_t)e * WID;
    for (int j = threadIdx.x; j < WID; j += 256)
        row[j] = f2bf(fmaxf(fmaf(d, w0[j], b0[j]), 0.0f));
}

// ---------------------------------------------------------------------------
// 3) Tiled bf16 WMMA GEMM:  Out[M,1024] = act( A[M,1024] @ BT^T + bias )
//    A row-major bf16, BT is N-major ([N][K]) bf16.
//    Block = 256 threads (8 waves) -> 128x128 output tile.
//    Wave (wm,wn): wm = wave&1 (2 along M, 64 rows), wn = wave>>1 (4 along N,
//    32 cols). 4 Mtiles x 2 Ntiles = 8 accumulators per wave.
//    LDS is double-buffered; tiles are filled with the CDNA5 async data
//    mover (global_load_async_to_lds_b128 / ASYNCcnt) so the fill of tile
//    k+1 overlaps the 8 WMMAs of tile k.
// ---------------------------------------------------------------------------
template <bool RELU>
__global__ __launch_bounds__(256) void gemm_bf16_kernel(
        const unsigned short* __restrict__ A,   // [M][1024]
        const unsigned short* __restrict__ BT,  // [1024][1024] N-major
        const float* __restrict__ bias,         // [1024]
        unsigned short* __restrict__ Out)       // [M][1024]
{
    __shared__ alignas(16) unsigned short Alds[2][128 * 32];   // 2 x 8 KB
    __shared__ alignas(16) unsigned short Blds[2][128 * 32];   // 2 x 8 KB

    const int tid  = threadIdx.x;
    const int lane = tid & 31;
    const int wave = tid >> 5;
    const int hf   = lane >> 4;       // K-half selector
    const int l15  = lane & 15;
    const int m0   = blockIdx.y * 128;
    const int n0   = blockIdx.x * 128;
    const int wm   = (wave & 1) * 64;
    const int wn   = (wave >> 1) * 32;

    // Per-thread staging coordinates: 512 16B chunks per 8 KB tile,
    // 256 threads x 2 passes.  row = chunk>>2, K-offset = (chunk&3)*8.
    const int r0  = tid >> 1;                    // pass 0: chunks 0..255
    const int c0  = (tid & 1) * 16;              // wait, see below
    (void)c0;

    auto stage = [&](int k0, int buf) {
#pragma unroll
        for (int p = 0; p < 2; ++p) {
            const int id  = p * 256 + tid;       // 0..511
            const int row = id >> 2;
            const int ck  = (id & 3) * 8;        // K element offset
            const unsigned short* gA = A  + (size_t)(m0 + row) * WID + k0 + ck;
            const unsigned short* gB = BT + (size_t)(n0 + row) * WID + k0 + ck;
            unsigned int la = (unsigned int)(uintptr_t)(&Alds[buf][row * 32 + ck]);
            unsigned int lb = (unsigned int)(uintptr_t)(&Blds[buf][row * 32 + ck]);
            async_g2l_b128(la, gA);
            async_g2l_b128(lb, gB);
        }
    };
    (void)r0;

    v8f acc[4][2];
#pragma unroll
    for (int mt = 0; mt < 4; ++mt)
#pragma unroll
        for (int nt = 0; nt < 2; ++nt)
#pragma unroll
            for (int r = 0; r < 8; ++r) acc[mt][nt][r] = 0.0f;

    stage(0, 0);                                  // prologue fill of buffer 0

    for (int k0 = 0; k0 < WID; k0 += 32) {
        const int buf = (k0 >> 5) & 1;

        // Publish this wave's async fills, then collective barrier: after it,
        // every wave's writes into Xlds[buf] are visible, and every wave has
        // finished reading the *other* buffer (so we may overwrite it).
        wait_async0();
        __syncthreads();

        if (k0 + 32 < WID) stage(k0 + 32, buf ^ 1);   // overlap next fill

        // ---- per-wave fragment loads (CDNA5 16-bit operand layout) ----
        FragBF aF[4], bF[2];
#pragma unroll
        for (int mt = 0; mt < 4; ++mt) {
            const unsigned short* base = &Alds[buf][(wm + mt * 16 + l15) * 32 + hf * 8];
            aF[mt].q[0] = *(const uint4*)(base);        // K = hf*8 .. +7
            aF[mt].q[1] = *(const uint4*)(base + 16);   // K = 16+hf*8 ..
        }
#pragma unroll
        for (int nt = 0; nt < 2; ++nt) {
            const unsigned short* base = &Blds[buf][(wn + nt * 16 + l15) * 32 + hf * 8];
            bF[nt].q[0] = *(const uint4*)(base);
            bF[nt].q[1] = *(const uint4*)(base + 16);
        }

#pragma unroll
        for (int mt = 0; mt < 4; ++mt)
#pragma unroll
            for (int nt = 0; nt < 2; ++nt)
                acc[mt][nt] = __builtin_amdgcn_wmma_f32_16x16x32_bf16(
                    false, aF[mt].v, false, bF[nt].v,
                    (short)0, acc[mt][nt], false, false);
    }

    // ---- epilogue: + bias, optional relu, store bf16 ----
#pragma unroll
    for (int nt = 0; nt < 2; ++nt) {
        const int n = n0 + wn + nt * 16 + l15;   // lane = N column
        const float bv = bias[n];
#pragma unroll
        for (int mt = 0; mt < 4; ++mt) {
#pragma unroll
            for (int r = 0; r < 8; ++r) {
                const int m = m0 + wm + mt * 16 + r + hf * 8;
                float v = acc[mt][nt][r] + bv;
                if (RELU) v = fmaxf(v, 0.0f);
                Out[(size_t)m * WID + n] = f2bf(v);
            }
        }
    }
}

// ---------------------------------------------------------------------------
// 4) dx_recon[b,o,:] = sum_e coeffs[b,e,o] * dist_vec[b,e,:]
//    grid (NATOMS/256, BATCH); dist_vec tile staged in LDS.
// ---------------------------------------------------------------------------
__global__ __launch_bounds__(256) void dx_kernel(const unsigned short* __restrict__ coeffs,
                                                 const float* __restrict__ dist_vec,
                                                 float* __restrict__ dx) {
    __shared__ float vec[EDGES * 3];             // 12 KB
    const int b = blockIdx.y;
    for (int i = threadIdx.x; i < EDGES * 3; i += 256)
        vec[i] = dist_vec[b * EDGES * 3 + i];
    __syncthreads();

    const int o = blockIdx.x * 256 + threadIdx.x;
    const unsigned short* cb = coeffs + (size_t)b * EDGES * NATOMS + o;
    float a0 = 0.f, a1 = 0.f, a2 = 0.f;
    for (int e = 0; e < EDGES; ++e) {
        float cf = bf2f(cb[(size_t)e * NATOMS]);
        a0 = fmaf(cf, vec[e * 3 + 0], a0);
        a1 = fmaf(cf, vec[e * 3 + 1], a1);
        a2 = fmaf(cf, vec[e * 3 + 2], a2);
    }
    const size_t gi = (size_t)(b * NATOMS + o) * 3;
    dx[gi + 0] = a0; dx[gi + 1] = a1; dx[gi + 2] = a2;
}

// ---------------------------------------------------------------------------
// 5) cg_offset[b,c,:] = sum_i dx[b,i,:] * assign_norm[b,i,c]
// ---------------------------------------------------------------------------
__global__ __launch_bounds__(128) void cgoff_kernel(const float* __restrict__ dx,
                                                    const float* __restrict__ an,
                                                    float* __restrict__ cgoff) {
    __shared__ float dxl[NATOMS * 3];            // 12 KB
    const int b = blockIdx.x;
    const int c = threadIdx.x;
    for (int i = c; i < NATOMS * 3; i += 128) dxl[i] = dx[b * NATOMS * 3 + i];
    __syncthreads();

    const float* ap = an + (size_t)b * NATOMS * NCGS + c;
    float a0 = 0.f, a1 = 0.f, a2 = 0.f;
    for (int i = 0; i < NATOMS; ++i) {
        float w = ap[(size_t)i * NCGS];
        a0 = fmaf(dxl[i * 3 + 0], w, a0);
        a1 = fmaf(dxl[i * 3 + 1], w, a1);
        a2 = fmaf(dxl[i * 3 + 2], w, a2);
    }
    const int gi = (b * NCGS + c) * 3;
    cgoff[gi + 0] = a0; cgoff[gi + 1] = a1; cgoff[gi + 2] = a2;
}

// ---------------------------------------------------------------------------
// 6) xyz_recon[b,i,:] = cg_xyz[b,idx,:] - cg_offset[b,idx,:] + dx[b,i,:]
// ---------------------------------------------------------------------------
__global__ __launch_bounds__(256) void recon_kernel(const float* __restrict__ cg_xyz,
                                                    const float* __restrict__ cgoff,
                                                    const float* __restrict__ dx,
                                                    const int* __restrict__ aidx,
                                                    float* __restrict__ out) {
    const int gi = blockIdx.x * 256 + threadIdx.x;   // 0..B*NATOMS-1
    const int b = gi >> 10;
    const int i = gi & 1023;
    const int c = aidx[i];
    const int s = (b * NCGS + c) * 3;
    const int d = gi * 3;
    out[d + 0] = cg_xyz[s + 0] - cgoff[s + 0] + dx[d + 0];
    out[d + 1] = cg_xyz[s + 1] - cgoff[s + 1] + dx[d + 1];
    out[d + 2] = cg_xyz[s + 2] - cgoff[s + 2] + dx[d + 2];
}

// ---------------------------------------------------------------------------
// Launch
// ---------------------------------------------------------------------------
extern "C" void kernel_launch(void* const* d_in, const int* in_sizes, int n_in,
                              void* d_out, int out_size, void* d_ws, size_t ws_size,
                              hipStream_t stream) {
    (void)in_sizes; (void)n_in; (void)out_size; (void)ws_size;

    const float* soft_assign = (const float*)d_in[0];
    const float* xyz         = (const float*)d_in[1];
    const float* cg_xyz      = (const float*)d_in[2];
    const float* assign_norm = (const float*)d_in[3];
    const int*   assign_idx  = (const int*)  d_in[4];
    const float* w0          = (const float*)d_in[5];
    const float* b0          = (const float*)d_in[6];
    const float* w1          = (const float*)d_in[7];
    const float* b1          = (const float*)d_in[8];
    const float* w2          = (const float*)d_in[9];
    const float* b2          = (const float*)d_in[10];
    float*       out         = (float*)d_out;

    // workspace carve-up (256B aligned)
    char* ws = (char*)d_ws;
    size_t off = 0;
    auto carve = [&](size_t bytes) {
        void* p = ws + off;
        off += (bytes + 255) & ~(size_t)255;
        return p;
    };
    unsigned short* w1T     = (unsigned short*)carve((size_t)WID * WID * 2);      // 2 MB
    unsigned short* w2T     = (unsigned short*)carve((size_t)WID * WID * 2);      // 2 MB
    unsigned short* h0buf   = (unsigned short*)carve((size_t)BE * WID * 2);       // 32 MB (reused for coeffs)
    unsigned short* h1buf   = (unsigned short*)carve((size_t)BE * WID * 2);       // 32 MB
    float*          distvec = (float*)carve((size_t)BE * 3 * 4);
    float*          distbuf = (float*)carve((size_t)BE * 4);
    float*          dxbuf   = (float*)carve((size_t)BATCH * NATOMS * 3 * 4);
    float*          cgoff   = (float*)carve((size_t)BATCH * NCGS * 3 * 4);

    // passthrough outputs: soft_assign, xyz
    const size_t n_sa  = (size_t)BATCH * NATOMS * NCGS;   // 2097152
    const size_t n_xyz = (size_t)BATCH * NATOMS * 3;      // 49152
    hipMemcpyAsync(out,        soft_assign, n_sa  * 4, hipMemcpyDeviceToDevice, stream);
    hipMemcpyAsync(out + n_sa, xyz,         n_xyz * 4, hipMemcpyDeviceToDevice, stream);
    float* out_recon = out + n_sa + n_xyz;

    wt_kernel <<<(WID * WID) / 256, 256, 0, stream>>>(w1, w2, w1T, w2T);
    knn_kernel<<<BATCH, 128, 0, stream>>>(cg_xyz, distvec, distbuf);
    h0_kernel <<<BE, 256, 0, stream>>>(distbuf, w0, b0, h0buf);

    dim3 ggrid(WID / 128, BE / 128);   // (8, 128)
    gemm_bf16_kernel<true ><<<ggrid, 256, 0, stream>>>(h0buf, w1T, b1, h1buf);
    gemm_bf16_kernel<false><<<ggrid, 256, 0, stream>>>(h1buf, w2T, b2, h0buf /*coeffs*/);

    dx_kernel   <<<dim3(NATOMS / 256, BATCH), 256, 0, stream>>>(h0buf, distvec, dxbuf);
    cgoff_kernel<<<BATCH, 128, 0, stream>>>(dxbuf, assign_norm, cgoff);
    recon_kernel<<<(BATCH * NATOMS) / 256, 256, 0, stream>>>(cg_xyz, cgoff, dxbuf, assign_idx, out_recon);
}
